// MyModel4_75557064671889
// MI455X (gfx1250) — matrix operand order repaired and checked by reading, not compiled, not run on hardware
//
#include <hip/hip_runtime.h>

#define N_PART 65536
#define N_EDGE 524288
#define NF 64
#define POS 3

typedef __bf16 bf16;
typedef __attribute__((ext_vector_type(16))) __bf16 v16bf;
typedef __attribute__((ext_vector_type(8)))  float  v8f;

// ---------------------------------------------------------------------------
// WMMA helpers (CDNA5 wave32, 16x16x32 bf16 -> f32)
// ---------------------------------------------------------------------------
__device__ __forceinline__ v8f wmma_bf16(v16bf a, v16bf b, v8f c) {
    return __builtin_amdgcn_wmma_f32_16x16x32_bf16(false, a, false, b, (short)0, c,
                                                   false, false);
}

// A fragment: 16x32 bf16 tile from LDS, row-major with leading dim `lda`.
__device__ __forceinline__ v16bf load_a_frag(const bf16* lds, int row_base, int lda,
                                             int k0, int lane) {
    const int r = lane & 15, hi = lane >> 4;
    const bf16* p = lds + (size_t)(row_base + r) * lda + k0;
    v16bf a;
#pragma unroll
    for (int e = 0; e < 8; ++e) a[e] = p[hi * 8 + e];
#pragma unroll
    for (int e = 0; e < 8; ++e) a[8 + e] = p[16 + hi * 8 + e];
    return a;
}

// B fragment: 32x16 tile; weights staged TRANSPOSED in LDS: wT[n * Kpad + k].
__device__ __forceinline__ v16bf load_b_fragT(const bf16* ldsT, int Kpad, int k0,
                                              int n0, int lane) {
    const int c = lane & 15, hi = lane >> 4;
    const bf16* p = ldsT + (size_t)(n0 + c) * Kpad + k0;
    v16bf b;
#pragma unroll
    for (int e = 0; e < 8; ++e) b[e] = p[hi * 8 + e];
#pragma unroll
    for (int e = 0; e < 8; ++e) b[8 + e] = p[16 + hi * 8 + e];
    return b;
}

__device__ __forceinline__ void stage_weightsT(bf16* dst, const float* w, int K, int Kpad,
                                               int Ncols, int Nsrc, int tid, int nthreads) {
    for (int i = tid; i < Ncols * Kpad; i += nthreads) {
        int n = i / Kpad, k = i - n * Kpad;
        float v = (k < K && n < Nsrc) ? w[(size_t)k * Nsrc + n] : 0.f;
        dst[i] = (bf16)v;
    }
}

__device__ __forceinline__ void stage_bias(float* dst, const float* b, int n, int nsrc,
                                           int tid, int nthreads) {
    for (int i = tid; i < n; i += nthreads) dst[i] = (i < nsrc) ? b[i] : 0.f;
}

// C/D layout: vgpr v, lanes 0-15 -> row v, col lane; lanes 16-31 -> row v+8, col lane-16.
__device__ __forceinline__ void epilogue_lds_relu(bf16* dst, int row_base, int ld, int lane,
                                                  const v8f* acc, const float* bias) {
    const int c = lane & 15, hi = lane >> 4;
#pragma unroll
    for (int t = 0; t < 4; ++t) {
        const int col = t * 16 + c;
        const float bv = bias[col];
#pragma unroll
        for (int v = 0; v < 8; ++v) {
            float x = fmaxf(acc[t][v] + bv, 0.f);
            dst[(size_t)(row_base + v + 8 * hi) * ld + col] = (bf16)x;
        }
    }
}

__device__ __forceinline__ void epilogue_global_bf16_relu(bf16* dst, int grow_base, int lane,
                                                          const v8f* acc, const float* bias) {
    const int c = lane & 15, hi = lane >> 4;
#pragma unroll
    for (int t = 0; t < 4; ++t) {
        const int col = t * 16 + c;
        const float bv = bias[col];
#pragma unroll
        for (int v = 0; v < 8; ++v) {
            float x = fmaxf(acc[t][v] + bv, 0.f);
            dst[(size_t)(grow_base + v + 8 * hi) * NF + col] = (bf16)x;
        }
    }
}

// ---------------------------------------------------------------------------
// CDNA5 async global->LDS copy (ASYNCcnt path, CDNA5 ISA)
// One aligned B128 per lane; LDS address taken from the VDST VGPR, GV addressing.
// ---------------------------------------------------------------------------
__device__ __forceinline__ void async_ld_b128(const void* gsrc, void* ldst) {
    unsigned lds_off = (unsigned)(unsigned long long)(uintptr_t)ldst;  // addr[31:0] = LDS offset
    unsigned long long gaddr = (unsigned long long)(uintptr_t)gsrc;
    asm volatile("global_load_async_to_lds_b128 %0, %1, off"
                 :: "v"(lds_off), "v"(gaddr) : "memory");
}
__device__ __forceinline__ void wait_async_le4() {
    asm volatile("s_wait_asynccnt 4" ::: "memory");
}
__device__ __forceinline__ void wait_async_0() {
    asm volatile("s_wait_asynccnt 0" ::: "memory");
}

// ---------------------------------------------------------------------------
// Zero-fill kernels (graph-capture safe scratch init)
// ---------------------------------------------------------------------------
__global__ void k_zero_f32(float* p, int n) {
    int i = blockIdx.x * blockDim.x + threadIdx.x;
    if (i < n) p[i] = 0.f;
}
__global__ void k_zero_bf16(bf16* p, int n) {
    int i = blockIdx.x * blockDim.x + threadIdx.x;
    if (i < n) p[i] = (bf16)0.f;
}

// ---------------------------------------------------------------------------
// 1) Particle encoder: relu(relu([attr,0,state] @ pe_w0 + b0) @ pe_w1 + b1)
// ---------------------------------------------------------------------------
__global__ void __launch_bounds__(256)
k_particle_enc(const float* __restrict__ attr, const float* __restrict__ state,
               const float* __restrict__ w0, const float* __restrict__ b0,
               const float* __restrict__ w1, const float* __restrict__ b1,
               bf16* __restrict__ out) {
    __shared__ __align__(16) bf16 sW0[64 * 32];   // [n][k], K=21 padded to 32
    __shared__ __align__(16) bf16 sW1[64 * 64];
    __shared__ float sB0[64], sB1[64];
    __shared__ __align__(16) bf16 sIn[128 * 32];
    __shared__ __align__(16) bf16 sH[128 * 64];

    const int tid = threadIdx.x;
    const int base = blockIdx.x * 128;

    stage_weightsT(sW0, w0, 21, 32, 64, 64, tid, 256);
    stage_weightsT(sW1, w1, 64, 64, 64, 64, tid, 256);
    stage_bias(sB0, b0, 64, 64, tid, 256);
    stage_bias(sB1, b1, 64, 64, tid, 256);

    // input row = [attr(9), zeros(6), state(6), pad(11)]
    for (int i = tid; i < 128 * 32; i += 256) {
        int r = i >> 5, c = i & 31;
        int gi = base + r;
        float v = 0.f;
        if (c < 9) v = attr[(size_t)gi * 9 + c];
        else if (c >= 15 && c < 21) v = state[(size_t)gi * 6 + (c - 15)];
        sIn[i] = (bf16)v;
    }
    __syncthreads();

    const int lane = tid & 31, row_base = (tid >> 5) * 16;

    v8f acc[4] = {};
    {
        v16bf a = load_a_frag(sIn, row_base, 32, 0, lane);
#pragma unroll
        for (int t = 0; t < 4; ++t)
            acc[t] = wmma_bf16(a, load_b_fragT(sW0, 32, 0, t * 16, lane), acc[t]);
    }
    epilogue_lds_relu(sH, row_base, 64, lane, acc, sB0);

    v8f acc2[4] = {};
#pragma unroll
    for (int k0 = 0; k0 < 64; k0 += 32) {
        v16bf a = load_a_frag(sH, row_base, 64, k0, lane);
#pragma unroll
        for (int t = 0; t < 4; ++t)
            acc2[t] = wmma_bf16(a, load_b_fragT(sW1, 64, k0, t * 16, lane), acc2[t]);
    }
    epilogue_global_bf16_relu(out, base + row_base, lane, acc2, sB1);
}

// ---------------------------------------------------------------------------
// 2) Relation encoder: 3x (Linear+ReLU) on gathered 43-wide edge rows
// ---------------------------------------------------------------------------
__global__ void __launch_bounds__(256)
k_relation_enc(const float* __restrict__ attr, const float* __restrict__ state,
               const float* __restrict__ Ra,
               const int* __restrict__ recv, const int* __restrict__ send,
               const float* __restrict__ w0, const float* __restrict__ b0,
               const float* __restrict__ w1, const float* __restrict__ b1,
               const float* __restrict__ w2, const float* __restrict__ b2,
               bf16* __restrict__ out) {
    __shared__ __align__(16) bf16 sW0[64 * 64];   // K=43 padded to 64
    __shared__ __align__(16) bf16 sW1[64 * 64];
    __shared__ __align__(16) bf16 sW2[64 * 64];
    __shared__ float sB0[64], sB1[64], sB2[64];
    __shared__ __align__(16) bf16 sIn[128 * 64];
    __shared__ __align__(16) bf16 sH[128 * 64];
    __shared__ int sR[128], sS[128];

    const int tid = threadIdx.x;
    const int base = blockIdx.x * 128;

    stage_weightsT(sW0, w0, 43, 64, 64, 64, tid, 256);
    stage_weightsT(sW1, w1, 64, 64, 64, 64, tid, 256);
    stage_weightsT(sW2, w2, 64, 64, 64, 64, tid, 256);
    stage_bias(sB0, b0, 64, 64, tid, 256);
    stage_bias(sB1, b1, 64, 64, tid, 256);
    stage_bias(sB2, b2, 64, 64, tid, 256);
    if (tid < 128) { sR[tid] = recv[base + tid]; sS[tid] = send[base + tid]; }
    __syncthreads();

    // row = [attr_r(9),0(6), attr_s(9),0(6), state_r(6), state_s(6), Ra(1), pad(21)]
    for (int i = tid; i < 128 * 64; i += 256) {
        int r = i >> 6, c = i & 63;
        int e = base + r, ri = sR[r], si = sS[r];
        float v = 0.f;
        if (c < 9) v = attr[(size_t)ri * 9 + c];
        else if (c >= 15 && c < 24) v = attr[(size_t)si * 9 + (c - 15)];
        else if (c >= 30 && c < 36) v = state[(size_t)ri * 6 + (c - 30)];
        else if (c >= 36 && c < 42) v = state[(size_t)si * 6 + (c - 36)];
        else if (c == 42) v = Ra[e];
        sIn[i] = (bf16)v;
    }
    __syncthreads();

    const int lane = tid & 31, row_base = (tid >> 5) * 16;

    v8f a0[4] = {};
#pragma unroll
    for (int k0 = 0; k0 < 64; k0 += 32) {
        v16bf a = load_a_frag(sIn, row_base, 64, k0, lane);
#pragma unroll
        for (int t = 0; t < 4; ++t)
            a0[t] = wmma_bf16(a, load_b_fragT(sW0, 64, k0, t * 16, lane), a0[t]);
    }
    epilogue_lds_relu(sH, row_base, 64, lane, a0, sB0);

    v8f a1[4] = {};
#pragma unroll
    for (int k0 = 0; k0 < 64; k0 += 32) {
        v16bf a = load_a_frag(sH, row_base, 64, k0, lane);
#pragma unroll
        for (int t = 0; t < 4; ++t)
            a1[t] = wmma_bf16(a, load_b_fragT(sW1, 64, k0, t * 16, lane), a1[t]);
    }
    epilogue_lds_relu(sIn, row_base, 64, lane, a1, sB1);

    v8f a2[4] = {};
#pragma unroll
    for (int k0 = 0; k0 < 64; k0 += 32) {
        v16bf a = load_a_frag(sIn, row_base, 64, k0, lane);
#pragma unroll
        for (int t = 0; t < 4; ++t)
            a2[t] = wmma_bf16(a, load_b_fragT(sW2, 64, k0, t * 16, lane), a2[t]);
    }
    epilogue_global_bf16_relu(out, base + row_base, lane, a2, sB2);
}

// ---------------------------------------------------------------------------
// 3) Edge propagation with async double-buffered staging:
//    rel_eff = relu([rel_enc, eff_r, eff_s] @ rp_w + rp_b); segment-sum via atomics.
//    K=192 in 3 chunks of 64; chunk i+1 streams in (ASYNCcnt) while i computes.
// ---------------------------------------------------------------------------
__device__ __forceinline__ void edge_compute_chunk(const bf16* buf, const bf16* sW,
                                                   int kbase, int row_base, int lane,
                                                   v8f* acc) {
#pragma unroll
    for (int kk = 0; kk < 64; kk += 32) {
        v16bf a = load_a_frag(buf, row_base, 64, kk, lane);
#pragma unroll
        for (int t = 0; t < 4; ++t)
            acc[t] = wmma_bf16(a, load_b_fragT(sW, 192, kbase + kk, t * 16, lane), acc[t]);
    }
}

__global__ void __launch_bounds__(256)
k_edge_prop(const bf16* __restrict__ rel_enc, const bf16* __restrict__ eff,
            const int* __restrict__ recv, const int* __restrict__ send,
            const float* __restrict__ rp_w, const float* __restrict__ rp_b,
            float* __restrict__ agg) {
    __shared__ __align__(16) bf16 sW[64 * 192];        // 24 KB, transposed [n][k]
    __shared__ __align__(16) bf16 sBuf[2][128 * 64];   // 2 x 16 KB chunk buffers
    __shared__ float sB[64];
    __shared__ int sR[128], sS[128];

    const int tid = threadIdx.x;
    const int base = blockIdx.x * 128;

    // chunk 0 (rel_enc rows, contiguous): 4 async B128 per lane, overlap weight staging
#pragma unroll
    for (int t = 0; t < 4; ++t) {
        int j = tid + t * 256;                 // 0..1023 transfers
        int r = j >> 3, s8 = (j & 7) * 8;      // row, 16B segment
        async_ld_b128(rel_enc + (size_t)(base + r) * NF + s8, &sBuf[0][r * 64 + s8]);
    }

    stage_weightsT(sW, rp_w, 192, 192, 64, 64, tid, 256);
    stage_bias(sB, rp_b, 64, 64, tid, 256);
    if (tid < 128) { sR[tid] = recv[base + tid]; sS[tid] = send[base + tid]; }
    __syncthreads();                            // indices + weights visible

    // chunk 1 (gathered eff[recv])
#pragma unroll
    for (int t = 0; t < 4; ++t) {
        int j = tid + t * 256;
        int r = j >> 3, s8 = (j & 7) * 8;
        async_ld_b128(eff + (size_t)sR[r] * NF + s8, &sBuf[1][r * 64 + s8]);
    }

    const int lane = tid & 31, row_base = (tid >> 5) * 16;
    v8f acc[4] = {};

    wait_async_le4();                           // own chunk-0 loads complete (in order)
    __syncthreads();                            // everyone's chunk-0 landed
    edge_compute_chunk(sBuf[0], sW, 0, row_base, lane, acc);
    __syncthreads();                            // all done reading sBuf[0]

    // chunk 2 (gathered eff[send]) -> reuse sBuf[0]
#pragma unroll
    for (int t = 0; t < 4; ++t) {
        int j = tid + t * 256;
        int r = j >> 3, s8 = (j & 7) * 8;
        async_ld_b128(eff + (size_t)sS[r] * NF + s8, &sBuf[0][r * 64 + s8]);
    }

    wait_async_le4();                           // own chunk-1 loads complete
    __syncthreads();
    edge_compute_chunk(sBuf[1], sW, 64, row_base, lane, acc);
    __syncthreads();

    wait_async_0();                             // own chunk-2 loads complete
    __syncthreads();
    edge_compute_chunk(sBuf[0], sW, 128, row_base, lane, acc);

    // bias + relu + scatter-add (segment_sum over recv)
    const int c = lane & 15, hi = lane >> 4;
#pragma unroll
    for (int t = 0; t < 4; ++t) {
        const int col = t * 16 + c;
        const float bv = sB[col];
#pragma unroll
        for (int v = 0; v < 8; ++v) {
            const int row = row_base + v + 8 * hi;
            float x = fmaxf(acc[t][v] + bv, 0.f);
            atomicAdd(&agg[(size_t)sR[row] * NF + col], x);
        }
    }
}

// ---------------------------------------------------------------------------
// 4) Particle update: eff = relu([particle_enc, agg] @ pp_w + pp_b)
//    penc half streamed async; f32 agg half converted in VALU (overlapped).
// ---------------------------------------------------------------------------
__global__ void __launch_bounds__(256)
k_particle_update(const bf16* __restrict__ penc, const float* __restrict__ agg,
                  const float* __restrict__ pp_w, const float* __restrict__ pp_b,
                  bf16* __restrict__ eff) {
    __shared__ __align__(16) bf16 sW[64 * 128];   // 16 KB
    __shared__ __align__(16) bf16 sIn[128 * 128]; // 32 KB
    __shared__ float sB[64];

    const int tid = threadIdx.x;
    const int base = blockIdx.x * 128;

    // async: penc -> sIn cols [0,64)
#pragma unroll
    for (int t = 0; t < 4; ++t) {
        int j = tid + t * 256;
        int r = j >> 3, s8 = (j & 7) * 8;
        async_ld_b128(penc + (size_t)(base + r) * NF + s8, &sIn[r * 128 + s8]);
    }

    stage_weightsT(sW, pp_w, 128, 128, 64, 64, tid, 256);
    stage_bias(sB, pp_b, 64, 64, tid, 256);

    // agg (f32 -> bf16) -> sIn cols [64,128)
    for (int i = tid; i < 128 * 64; i += 256) {
        int r = i >> 6, c = i & 63;
        sIn[r * 128 + 64 + c] = (bf16)agg[(size_t)(base + r) * NF + c];
    }
    wait_async_0();
    __syncthreads();

    const int lane = tid & 31, row_base = (tid >> 5) * 16;

    v8f acc[4] = {};
#pragma unroll
    for (int k0 = 0; k0 < 128; k0 += 32) {
        v16bf a = load_a_frag(sIn, row_base, 128, k0, lane);
#pragma unroll
        for (int t = 0; t < 4; ++t)
            acc[t] = wmma_bf16(a, load_b_fragT(sW, 128, k0, t * 16, lane), acc[t]);
    }
    epilogue_global_bf16_relu(eff, base + row_base, lane, acc, sB);
}

// ---------------------------------------------------------------------------
// 5) Predictor: relu(l0), relu(l1), l2 (64 -> 64 -> 64 -> 3), async input stage
// ---------------------------------------------------------------------------
__global__ void __launch_bounds__(256)
k_predictor(const bf16* __restrict__ eff,
            const float* __restrict__ w0, const float* __restrict__ b0,
            const float* __restrict__ w1, const float* __restrict__ b1,
            const float* __restrict__ w2, const float* __restrict__ b2,
            float* __restrict__ out) {
    __shared__ __align__(16) bf16 sW0[64 * 64];
    __shared__ __align__(16) bf16 sW1[64 * 64];
    __shared__ __align__(16) bf16 sW2[16 * 64];   // N padded 3 -> 16, transposed
    __shared__ float sB0[64], sB1[64], sB2[16];
    __shared__ __align__(16) bf16 sIn[128 * 64];
    __shared__ __align__(16) bf16 sH[128 * 64];

    const int tid = threadIdx.x;
    const int base = blockIdx.x * 128;

    // async input stage (contiguous bf16 rows)
#pragma unroll
    for (int t = 0; t < 4; ++t) {
        int j = tid + t * 256;
        int r = j >> 3, s8 = (j & 7) * 8;
        async_ld_b128(eff + (size_t)(base + r) * NF + s8, &sIn[r * 64 + s8]);
    }

    stage_weightsT(sW0, w0, 64, 64, 64, 64, tid, 256);
    stage_weightsT(sW1, w1, 64, 64, 64, 64, tid, 256);
    stage_weightsT(sW2, w2, 64, 64, 16, 3, tid, 256);
    stage_bias(sB0, b0, 64, 64, tid, 256);
    stage_bias(sB1, b1, 64, 64, tid, 256);
    stage_bias(sB2, b2, 16, 3, tid, 256);
    wait_async_0();
    __syncthreads();

    const int lane = tid & 31, row_base = (tid >> 5) * 16;

    v8f a0[4] = {};
#pragma unroll
    for (int k0 = 0; k0 < 64; k0 += 32) {
        v16bf a = load_a_frag(sIn, row_base, 64, k0, lane);
#pragma unroll
        for (int t = 0; t < 4; ++t)
            a0[t] = wmma_bf16(a, load_b_fragT(sW0, 64, k0, t * 16, lane), a0[t]);
    }
    epilogue_lds_relu(sH, row_base, 64, lane, a0, sB0);

    v8f a1[4] = {};
#pragma unroll
    for (int k0 = 0; k0 < 64; k0 += 32) {
        v16bf a = load_a_frag(sH, row_base, 64, k0, lane);
#pragma unroll
        for (int t = 0; t < 4; ++t)
            a1[t] = wmma_bf16(a, load_b_fragT(sW1, 64, k0, t * 16, lane), a1[t]);
    }
    epilogue_lds_relu(sIn, row_base, 64, lane, a1, sB1);

    v8f a2 = {};
#pragma unroll
    for (int k0 = 0; k0 < 64; k0 += 32) {
        v16bf a = load_a_frag(sIn, row_base, 64, k0, lane);
        a2 = wmma_bf16(a, load_b_fragT(sW2, 64, k0, 0, lane), a2);
    }
    const int c = lane & 15, hi = lane >> 4;
    if (c < POS) {
        const float bv = sB2[c];
#pragma unroll
        for (int v = 0; v < 8; ++v) {
            const int row = base + row_base + v + 8 * hi;
            out[(size_t)row * POS + c] = a2[v] + bv;
        }
    }
}

// ---------------------------------------------------------------------------
// Launch
// ---------------------------------------------------------------------------
extern "C" void kernel_launch(void* const* d_in, const int* in_sizes, int n_in,
                              void* d_out, int out_size, void* d_ws, size_t ws_size,
                              hipStream_t stream) {
    (void)in_sizes; (void)n_in; (void)out_size; (void)ws_size;

    const float* attr  = (const float*)d_in[0];
    const float* state = (const float*)d_in[1];
    const float* Ra    = (const float*)d_in[2];
    const int*   recv  = (const int*)d_in[3];
    const int*   send  = (const int*)d_in[4];
    const float* pe_w0 = (const float*)d_in[5],  *pe_b0 = (const float*)d_in[6];
    const float* pe_w1 = (const float*)d_in[7],  *pe_b1 = (const float*)d_in[8];
    const float* re_w0 = (const float*)d_in[9],  *re_b0 = (const float*)d_in[10];
    const float* re_w1 = (const float*)d_in[11], *re_b1 = (const float*)d_in[12];
    const float* re_w2 = (const float*)d_in[13], *re_b2 = (const float*)d_in[14];
    const float* rp_w  = (const float*)d_in[15], *rp_b  = (const float*)d_in[16];
    const float* pp_w  = (const float*)d_in[17], *pp_b  = (const float*)d_in[18];
    const float* fp_w0 = (const float*)d_in[19], *fp_b0 = (const float*)d_in[20];
    const float* fp_w1 = (const float*)d_in[21], *fp_b1 = (const float*)d_in[22];
    const float* fp_w2 = (const float*)d_in[23], *fp_b2 = (const float*)d_in[24];
    float* out = (float*)d_out;

    char* ws = (char*)d_ws;
    size_t off = 0;
    bf16* rel_enc = (bf16*)(ws + off); off += (size_t)N_EDGE * NF * sizeof(bf16);
    bf16* penc    = (bf16*)(ws + off); off += (size_t)N_PART * NF * sizeof(bf16);
    bf16* eff     = (bf16*)(ws + off); off += (size_t)N_PART * NF * sizeof(bf16);
    float* agg    = (float*)(ws + off); off += (size_t)N_PART * NF * sizeof(float);

    const int nNF = N_PART * NF;                    // 4,194,304
    const int zblk = (nNF + 255) / 256;

    k_zero_bf16<<<zblk, 256, 0, stream>>>(eff, nNF);          // effect_0 = 0
    k_particle_enc<<<N_PART / 128, 256, 0, stream>>>(attr, state, pe_w0, pe_b0,
                                                     pe_w1, pe_b1, penc);
    k_relation_enc<<<N_EDGE / 128, 256, 0, stream>>>(attr, state, Ra, recv, send,
                                                     re_w0, re_b0, re_w1, re_b1,
                                                     re_w2, re_b2, rel_enc);
    for (int step = 0; step < 2; ++step) {
        k_zero_f32<<<zblk, 256, 0, stream>>>(agg, nNF);
        k_edge_prop<<<N_EDGE / 128, 256, 0, stream>>>(rel_enc, eff, recv, send,
                                                      rp_w, rp_b, agg);
        k_particle_update<<<N_PART / 128, 256, 0, stream>>>(penc, agg, pp_w, pp_b, eff);
    }
    k_predictor<<<N_PART / 128, 256, 0, stream>>>(eff, fp_w0, fp_b0, fp_w1, fp_b1,
                                                  fp_w2, fp_b2, out);
}
